// MemoryMHA_70377334113036
// MI455X (gfx1250) — matrix operand order repaired and checked by reading, not compiled, not run on hardware
//
#include <hip/hip_runtime.h>

// ---------------- problem constants ----------------
#define B_BATCH 4096
#define DIN     768
#define S_TOK   17
#define M_TOK   8
#define T_TOK   25          // S + M
#define TP      32          // tokens padded to 2 WMMA M-tiles
#define KB      24          // 768 / 32 k-steps per WMMA chain
#define NT      48          // 768 / 16 output-column tiles
#define NB      24          // 768 / 32 -> 2-wide nt blocks

typedef float  v8f     __attribute__((ext_vector_type(8)));
typedef __bf16 bf16x16 __attribute__((ext_vector_type(16)));
typedef __bf16 bf16x8  __attribute__((ext_vector_type(8)));
typedef __bf16 bf16x4  __attribute__((ext_vector_type(4)));

#define WMMA_BF16(a, b, c) \
    __builtin_amdgcn_wmma_f32_16x16x32_bf16(false, (a), false, (b), (short)0, (c), false, false)

// A-operand (16x32 bf16): lane holds row m = lane%16; elements 0..15 -> K = k0+16g .. +15
// (contiguous, matching B). Two b128 LDS loads.
__device__ __forceinline__ bf16x16 load_a16(const __bf16* p) {
    bf16x8 lo = *(const bf16x8*)p;
    bf16x8 hi = *(const bf16x8*)(p + 8);
    bf16x16 a;
#pragma unroll
    for (int i = 0; i < 8; ++i) { a[i] = lo[i]; a[i + 8] = hi[i]; }
    return a;
}

// ---------------- weight f32 -> bf16 conversion (runs once per launch) ----------------
__global__ __launch_bounds__(256) void wcvt_kernel(const float* __restrict__ wq,
                                                   const float* __restrict__ wk,
                                                   const float* __restrict__ wv,
                                                   const float* __restrict__ wo,
                                                   __bf16* __restrict__ dst) {
    const int N = DIN * DIN;
    int i = blockIdx.x * 256 + threadIdx.x;           // 0 .. 4N-1
    const float* src = (i < N) ? wq : (i < 2 * N) ? wk : (i < 3 * N) ? wv : wo;
    dst[i] = (__bf16)src[i % N];
}

// ---------------- fused memory-MHA kernel: one workgroup per batch ----------------
__global__ __launch_bounds__(256)
void MemoryMHA_fused_kernel(const float* __restrict__ x,     // [B,17,768]
                            const float* __restrict__ mem,   // [8,768]
                            const float* __restrict__ bq,
                            const float* __restrict__ bk,
                            const float* __restrict__ bv,
                            const float* __restrict__ bo,
                            const __bf16* __restrict__ wbf,  // [4][768][768] bf16: q,k,v,o
                            float* __restrict__ out) {       // [B,17,768]
    extern __shared__ char smem[];
    __bf16* c_lds  = (__bf16*)smem;                 // [32][768]
    __bf16* q_lds  = c_lds + TP * DIN;              // [32][768]  (reused as head_out)
    __bf16* k_lds  = q_lds + TP * DIN;              // [32][768]
    __bf16* vt_lds = k_lds + TP * DIN;              // [768][32]  (V transposed)
    float*  s_lds  = (float*)(vt_lds + DIN * TP);   // [32][32] raw scores
    __bf16* a_lds  = (__bf16*)(s_lds + TP * TP);    // [32][32] softmax probs (bf16)

    const int tid  = threadIdx.x;
    const int wave = tid >> 5;
    const int lane = tid & 31;
    const int g    = lane >> 4;   // lane group 0/1
    const int ln16 = lane & 15;
    const int b    = blockIdx.x;

    // ---- stage 0: build concat(x, memory) as bf16 in LDS; pad rows 25..31 with 0 ----
    {
        const float4* xb = (const float4*)(x + (size_t)b * S_TOK * DIN);
        const float4* mb = (const float4*)mem;
        for (int v4 = tid; v4 < TP * DIN / 4; v4 += 256) {
            int t   = (v4 * 4) / DIN;
            int col = (v4 * 4) % DIN;
            float4 f;
            if (t < S_TOK)       f = xb[(t * DIN + col) >> 2];
            else if (t < T_TOK)  f = mb[((t - S_TOK) * DIN + col) >> 2];
            else                 f = make_float4(0.f, 0.f, 0.f, 0.f);
            bf16x4 h;
            h[0] = (__bf16)f.x; h[1] = (__bf16)f.y; h[2] = (__bf16)f.z; h[3] = (__bf16)f.w;
            *(bf16x4*)(&c_lds[t * DIN + col]) = h;
        }
    }
    __syncthreads();

    // ---- stage 1: Q/K/V projections with 2x2 register blocking ----
    // 3 GEMMs x 24 blocks (each block = mt{0,1} x nt{2*nb, 2*nb+1}) = 72 blocks / 8 waves.
    for (int blk = wave; blk < 3 * NB; blk += 8) {
        const int gm = blk / NB;              // 0=Q 1=K 2=V
        const int nb = blk % NB;
        const int nt0 = nb * 2;
        const __bf16* w = wbf + (size_t)gm * DIN * DIN;
        const __bf16* wr0 = w + (size_t)(nt0 * 16 + ln16) * DIN;       // B lane row (nt0)
        const __bf16* wr1 = wr0 + (size_t)16 * DIN;                    // B lane row (nt1)
        v8f acc00 = {}, acc01 = {}, acc10 = {}, acc11 = {};
#pragma unroll 2
        for (int kb = 0; kb < KB; ++kb) {
            const int ko = kb * 32;
            if (kb + 1 < KB)
                __builtin_prefetch(wr0 + ko + 32 + 16 * g, 0, 3);
            bf16x16 a0 = load_a16(&c_lds[(ln16)      * DIN + ko + 16 * g]);
            bf16x16 a1 = load_a16(&c_lds[(16 + ln16) * DIN + ko + 16 * g]);
            bf16x16 b0 = *(const bf16x16*)(wr0 + ko + 16 * g);
            bf16x16 b1 = *(const bf16x16*)(wr1 + ko + 16 * g);
            acc00 = WMMA_BF16(a0, b0, acc00);
            acc01 = WMMA_BF16(a0, b1, acc01);
            acc10 = WMMA_BF16(a1, b0, acc10);
            acc11 = WMMA_BF16(a1, b1, acc11);
        }
        const float* bp = (gm == 0) ? bq : (gm == 1) ? bk : bv;
        const float bias0 = bp[nt0 * 16 + ln16];
        const float bias1 = bp[nt0 * 16 + 16 + ln16];
        if (gm < 2) {
            __bf16* dst = (gm == 0) ? q_lds : k_lds;    // row-major [t][d]
#pragma unroll
            for (int r = 0; r < 8; ++r) {
                const int mr = g * 8 + r;               // D-layout: m = r + 8g (+16*mt)
                dst[mr * DIN        + nt0 * 16      + ln16] = (__bf16)(acc00[r] + bias0);
                dst[mr * DIN        + nt0 * 16 + 16 + ln16] = (__bf16)(acc01[r] + bias1);
                dst[(16 + mr) * DIN + nt0 * 16      + ln16] = (__bf16)(acc10[r] + bias0);
                dst[(16 + mr) * DIN + nt0 * 16 + 16 + ln16] = (__bf16)(acc11[r] + bias1);
            }
        } else {                                        // V stored transposed: [d][t]
#pragma unroll
            for (int r = 0; r < 8; ++r) {
                const int mr = g * 8 + r;
                vt_lds[(nt0 * 16      + ln16) * TP + mr]      = (__bf16)(acc00[r] + bias0);
                vt_lds[(nt0 * 16 + 16 + ln16) * TP + mr]      = (__bf16)(acc01[r] + bias1);
                vt_lds[(nt0 * 16      + ln16) * TP + 16 + mr] = (__bf16)(acc10[r] + bias0);
                vt_lds[(nt0 * 16 + 16 + ln16) * TP + 16 + mr] = (__bf16)(acc11[r] + bias1);
            }
        }
    }
    __syncthreads();

    // ---- stage 2: scores S = Q @ K^T  (4 16x16 tiles, waves 0..3 in parallel) ----
    if (wave < 4) {
        const int mt = wave >> 1, nt = wave & 1;
        const int m    = mt * 16 + ln16;
        const int nrow = nt * 16 + ln16;                // lane n reads K row n (K^T column)
        v8f acc = {};
#pragma unroll 4
        for (int kb = 0; kb < KB; ++kb) {
            bf16x16 a  = load_a16(&q_lds[m * DIN + kb * 32 + 16 * g]);
            bf16x16 bm = *(const bf16x16*)(&k_lds[nrow * DIN + kb * 32 + 16 * g]);
            acc = WMMA_BF16(a, bm, acc);
        }
#pragma unroll
        for (int r = 0; r < 8; ++r)
            s_lds[(mt * 16 + g * 8 + r) * TP + nt * 16 + ln16] = acc[r];
    }
    __syncthreads();

    // ---- stage 3: masked softmax, rows 0..16 over cols 0..24; rows >=17 zeroed ----
    if (tid < TP) {
        const int t = tid;
        if (t < S_TOK) {
            const float scale = 0.03608439182435161f;   // 1/sqrt(768)
            float row[T_TOK];
            float mx = -1e30f;
#pragma unroll
            for (int j = 0; j < T_TOK; ++j) {
                row[j] = s_lds[t * TP + j] * scale;
                mx = fmaxf(mx, row[j]);
            }
            float sum = 0.f;
#pragma unroll
            for (int j = 0; j < T_TOK; ++j) { row[j] = __expf(row[j] - mx); sum += row[j]; }
            const float inv = 1.f / sum;
#pragma unroll
            for (int j = 0; j < TP; ++j)
                a_lds[t * TP + j] = (__bf16)((j < T_TOK) ? row[j] * inv : 0.f);
        } else {
            for (int j = 0; j < TP; ++j) a_lds[t * TP + j] = (__bf16)0.f;
        }
    }
    __syncthreads();

    // ---- stage 4: head_out = attn @ V (one k-step, 2x2 blocks) -> overwrite q_lds ----
    for (int blk = wave; blk < NB; blk += 8) {
        const int nt0 = blk * 2;
        bf16x16 a0 = load_a16(&a_lds[(ln16)      * TP + 16 * g]);
        bf16x16 a1 = load_a16(&a_lds[(16 + ln16) * TP + 16 * g]);
        bf16x16 b0 = *(const bf16x16*)(&vt_lds[(nt0 * 16      + ln16) * TP + 16 * g]);
        bf16x16 b1 = *(const bf16x16*)(&vt_lds[(nt0 * 16 + 16 + ln16) * TP + 16 * g]);
        v8f acc00 = {}, acc01 = {}, acc10 = {}, acc11 = {};
        acc00 = WMMA_BF16(a0, b0, acc00);
        acc01 = WMMA_BF16(a0, b1, acc01);
        acc10 = WMMA_BF16(a1, b0, acc10);
        acc11 = WMMA_BF16(a1, b1, acc11);
#pragma unroll
        for (int r = 0; r < 8; ++r) {
            const int mr = g * 8 + r;
            q_lds[mr * DIN        + nt0 * 16      + ln16] = (__bf16)acc00[r];
            q_lds[mr * DIN        + nt0 * 16 + 16 + ln16] = (__bf16)acc01[r];
            q_lds[(16 + mr) * DIN + nt0 * 16      + ln16] = (__bf16)acc10[r];
            q_lds[(16 + mr) * DIN + nt0 * 16 + 16 + ln16] = (__bf16)acc11[r];
        }
    }
    __syncthreads();

    // ---- stage 5: out = head_out @ Wo^T + bo (2x2 blocks), store only rows < 17 ----
    {
        const __bf16* wo = wbf + (size_t)3 * DIN * DIN;
        float* ob = out + (size_t)b * S_TOK * DIN;
        for (int blk = wave; blk < NB; blk += 8) {
            const int nt0 = blk * 2;
            const __bf16* wr0 = wo + (size_t)(nt0 * 16 + ln16) * DIN;
            const __bf16* wr1 = wr0 + (size_t)16 * DIN;
            v8f acc00 = {}, acc01 = {}, acc10 = {}, acc11 = {};
#pragma unroll 2
            for (int kb = 0; kb < KB; ++kb) {
                const int ko = kb * 32;
                if (kb + 1 < KB)
                    __builtin_prefetch(wr0 + ko + 32 + 16 * g, 0, 3);
                bf16x16 a0 = load_a16(&q_lds[(ln16)      * DIN + ko + 16 * g]);
                bf16x16 a1 = load_a16(&q_lds[(16 + ln16) * DIN + ko + 16 * g]);
                bf16x16 b0 = *(const bf16x16*)(wr0 + ko + 16 * g);
                bf16x16 b1 = *(const bf16x16*)(wr1 + ko + 16 * g);
                acc00 = WMMA_BF16(a0, b0, acc00);
                acc01 = WMMA_BF16(a0, b1, acc01);
                acc10 = WMMA_BF16(a1, b0, acc10);
                acc11 = WMMA_BF16(a1, b1, acc11);
            }
            const float bias0 = bo[nt0 * 16 + ln16];
            const float bias1 = bo[nt0 * 16 + 16 + ln16];
#pragma unroll
            for (int r = 0; r < 8; ++r) {
                const int m0 = g * 8 + r;       // rows from mt=0 tiles (0..15): always kept
                ob[m0 * DIN + nt0 * 16      + ln16] = acc00[r] + bias0;
                ob[m0 * DIN + nt0 * 16 + 16 + ln16] = acc01[r] + bias1;
                const int m1 = 16 + g * 8 + r;  // rows from mt=1 tiles: only row 16 kept
                if (m1 < S_TOK) {
                    ob[m1 * DIN + nt0 * 16      + ln16] = acc10[r] + bias0;
                    ob[m1 * DIN + nt0 * 16 + 16 + ln16] = acc11[r] + bias1;
                }
            }
        }
    }
}

extern "C" void kernel_launch(void* const* d_in, const int* in_sizes, int n_in,
                              void* d_out, int out_size, void* d_ws, size_t ws_size,
                              hipStream_t stream) {
    (void)in_sizes; (void)n_in; (void)out_size; (void)ws_size;
    const float* x   = (const float*)d_in[0];
    const float* mem = (const float*)d_in[1];
    const float* wq  = (const float*)d_in[2];
    const float* bq  = (const float*)d_in[3];
    const float* wk  = (const float*)d_in[4];
    const float* bk  = (const float*)d_in[5];
    const float* wv  = (const float*)d_in[6];
    const float* bv  = (const float*)d_in[7];
    const float* wo  = (const float*)d_in[8];
    const float* bo  = (const float*)d_in[9];
    __bf16* wbf = (__bf16*)d_ws;                       // 4 * 768*768 bf16 = 4.7 MB

    const int N = DIN * DIN;
    wcvt_kernel<<<(4 * N) / 256, 256, 0, stream>>>(wq, wk, wv, wo, wbf);

    // LDS: c + q + k + vt (4 * 49152 B) + scores (4096 B) + attn (2048 B) = 202752 B
    const size_t lds_bytes = (size_t)4 * TP * DIN * sizeof(__bf16)
                           + (size_t)TP * TP * sizeof(float)
                           + (size_t)TP * TP * sizeof(__bf16);
    MemoryMHA_fused_kernel<<<B_BATCH, 256, lds_bytes, stream>>>(
        x, mem, bq, bk, bv, bo, wbf, (float*)d_out);
}
